// MLPLayer_90400471646564
// MI455X (gfx1250) — matrix operand order repaired and checked by reading, not compiled, not run on hardware
//
#include <hip/hip_runtime.h>
#include <cstddef>

// ---------------------------------------------------------------------------
// MLPLayer with contextual sparsity (predictor + top-k + gated MLP on 716 of
// 14336 channels).  Heavy work = 3 GEMMs (~72 GFLOP) via v_wmma_f32_16x16x32_bf16
// with double-buffered LDS tiles staged by GLOBAL_LOAD_ASYNC_TO_LDS_B128
// (ASYNCcnt) when the toolchain exposes it; sync ds_store staging otherwise.
// ---------------------------------------------------------------------------

typedef __attribute__((ext_vector_type(16))) __bf16        v16bf;
typedef __attribute__((ext_vector_type(8)))  float         v8f;
typedef __attribute__((ext_vector_type(4)))  unsigned int  u32x4;
typedef int b128v __attribute__((vector_size(16)));        // matches builtin's pointee

constexpr int HIDDEN = 4096;
constexpr int INTER  = 14336;
constexpr int PRED_H = 1024;
constexpr int HC     = 716;   // int(0.05 * 14336)
constexpr int KP     = 768;   // HC padded to multiple of 32/128 for WMMA tiles
constexpr int SEQ    = 4096;
constexpr int LDT    = 40;    // LDS tile row stride (bf16 elems) = 80B, 16B-aligned

#if defined(__HIP_DEVICE_COMPILE__) && \
    __has_builtin(__builtin_amdgcn_global_load_async_to_lds_b128) && \
    __has_builtin(__builtin_amdgcn_s_wait_asynccnt)
#define USE_ASYNC 1
#else
#define USE_ASYNC 0
#endif

__device__ inline unsigned short f2bf(float f) {        // fp32 -> bf16 RNE
  unsigned int u = __builtin_bit_cast(unsigned int, f);
  u += 0x7FFFu + ((u >> 16) & 1u);
  return (unsigned short)(u >> 16);
}

// stage one 16B chunk global -> LDS
__device__ inline void cp16(const void* g, void* l) {
#if USE_ASYNC
  __builtin_amdgcn_global_load_async_to_lds_b128(
      (__attribute__((address_space(1))) b128v*)(size_t)g,
      (__attribute__((address_space(3))) b128v*)(unsigned int)(size_t)l,
      0, 0);
#else
  *(u32x4*)l = *(const u32x4*)g;
#endif
}
__device__ inline void wait_stage() {
#if USE_ASYNC
  __builtin_amdgcn_s_wait_asynccnt(0);
#endif
}

union FragB { v16bf v; u32x4 q[2]; };   // 32B: one WMMA A or B fragment
union Acc   { v8f   v; float f[8]; };   // 32B: one 16x16 f32 C/D fragment

// --------------------------- predictor GEMVs -------------------------------
// p[j] = dot(x_last, W1[j,:])   (1024 outputs, len-4096) — one wave per output
__global__ void k_pred1(const float* __restrict__ x_last,
                        const float* __restrict__ W1, float* __restrict__ p) {
  int gw   = (blockIdx.x * 256 + threadIdx.x) >> 5;
  int lane = threadIdx.x & 31;
  float acc = 0.f;
  const float* wrow = W1 + (size_t)gw * HIDDEN;
  for (int k = lane; k < HIDDEN; k += 32) acc += x_last[k] * wrow[k];
  for (int m = 16; m; m >>= 1) acc += __shfl_xor(acc, m, 32);
  if (lane == 0) p[gw] = acc;
}

// h[i] = dot(p, W2[i,:])   (14336 outputs, len-1024) — one wave per output
__global__ void k_pred2(const float* __restrict__ p,
                        const float* __restrict__ W2, float* __restrict__ h) {
  int gw   = (blockIdx.x * 256 + threadIdx.x) >> 5;
  int lane = threadIdx.x & 31;
  float acc = 0.f;
  const float* wrow = W2 + (size_t)gw * PRED_H;
  for (int k = lane; k < PRED_H; k += 32) acc += p[k] * wrow[k];
  for (int m = 16; m; m >>= 1) acc += __shfl_xor(acc, m, 32);
  if (lane == 0) h[gw] = acc;
}

// ------------------------------- top-k -------------------------------------
// rank(i) = #{j : h[j] > h[i]  or  (h[j]==h[i] and j<i)}  (jax tie-break).
// Element i is selected iff rank < HC; idx[rank] = i gives descending order.
__global__ __launch_bounds__(256) void k_topk(const float* __restrict__ h,
                                              int* __restrict__ idx) {
  __shared__ float sh[INTER];
  for (int i = threadIdx.x; i < INTER; i += 256) sh[i] = h[i];
  __syncthreads();
  int i = blockIdx.x * 256 + threadIdx.x;          // grid covers 0..INTER-1
  if (i >= HC && i < KP) idx[i] = -1;              // pad slots
  float hi = sh[i];
  int rank = 0;
  for (int j = 0; j < INTER; ++j) {
    float hj = sh[j];
    rank += (hj > hi) || (hj == hi && j < i);
  }
  if (rank < HC) idx[rank] = i;
}

// --------------------- gathers + bf16 conversion ---------------------------
__global__ void k_cvtx(const float* __restrict__ x, unsigned short* __restrict__ xb) {
  size_t stride = (size_t)gridDim.x * blockDim.x;
  for (size_t i = blockIdx.x * (size_t)blockDim.x + threadIdx.x;
       i < (size_t)SEQ * HIDDEN; i += stride)
    xb[i] = f2bf(x[i]);
}

__global__ void k_gather_gu(const float* __restrict__ Wg, const float* __restrict__ Wu,
                            const int* __restrict__ idx,
                            unsigned short* __restrict__ WgS,
                            unsigned short* __restrict__ WuS) {
  int r = blockIdx.x;                  // 0..KP-1 (selected-channel row)
  int j = idx[r];                      // -1 for pad rows -> zero fill
  const float* gsrc = Wg + (size_t)(j < 0 ? 0 : j) * HIDDEN;
  const float* usrc = Wu + (size_t)(j < 0 ? 0 : j) * HIDDEN;
  for (int c = threadIdx.x; c < HIDDEN; c += 256) {
    unsigned short g = 0, u = 0;
    if (j >= 0) { g = f2bf(gsrc[c]); u = f2bf(usrc[c]); }
    WgS[(size_t)r * HIDDEN + c] = g;
    WuS[(size_t)r * HIDDEN + c] = u;
  }
}

// WdS[n][kp] = Wd[n, idx[kp]]  — [n][k] layout feeds B fragments directly
__global__ void k_gather_d(const float* __restrict__ Wd, const int* __restrict__ idx,
                           unsigned short* __restrict__ WdS) {
  int n = blockIdx.x;                  // 0..HIDDEN-1
  for (int kp = threadIdx.x; kp < KP; kp += 256) {
    int j = idx[kp];
    WdS[(size_t)n * KP + kp] = (j >= 0) ? f2bf(Wd[(size_t)n * INTER + j]) : 0;
  }
}

// -------------------- fused gate/up WMMA GEMM + SiLU -----------------------
// T[m][kp] = silu(x@WgSᵀ) * (x@WuSᵀ).  M=SEQ, N=KP, K=HIDDEN.
// Block = 256 thr = 8 waves, tile 128(M) x 64(N); wave -> 32x32 sub-tile,
// 2x2 fragments per matrix; K staged 32-wide, double-buffered LDS.
__global__ __launch_bounds__(256) void k_gateup(
    const unsigned short* __restrict__ X,    // SEQ x HIDDEN bf16
    const unsigned short* __restrict__ WgS,  // KP x HIDDEN bf16  ([n][k])
    const unsigned short* __restrict__ WuS,
    unsigned short* __restrict__ T) {        // SEQ x KP bf16
  __shared__ __align__(16) unsigned short As[2 * 128 * LDT];
  __shared__ __align__(16) unsigned short Bg[2 * 64 * LDT];
  __shared__ __align__(16) unsigned short Bu[2 * 64 * LDT];
  const int tid = threadIdx.x, lane = tid & 31, w = tid >> 5;
  const int wm = w & 3, wn = w >> 2;              // wave's 32x32 region
  const int m0 = blockIdx.x * 128, n0 = blockIdx.y * 64;

  Acc ag[2][2], au[2][2];
  for (int i = 0; i < 2; ++i)
    for (int j = 0; j < 2; ++j)
      for (int e = 0; e < 8; ++e) { ag[i][j].f[e] = 0.f; au[i][j].f[e] = 0.f; }

  const int ar = tid >> 2;        // staging rows/chunks (fixed per thread)
  const int ac = tid & 3;
  const int mrow = lane & 15, kh = (lane >> 4) * 8, kb = (lane >> 4) * 16;

  auto stage = [&](int buf, int kk) {
    const int ao = buf * 128 * LDT, bo = buf * 64 * LDT;
    #pragma unroll
    for (int t2 = 0; t2 < 2; ++t2) {             // A: 128x32 = 512 chunks
      int r = ar + t2 * 64;
      cp16(&X[(size_t)(m0 + r) * HIDDEN + kk + ac * 8], &As[ao + r * LDT + ac * 8]);
    }
    cp16(&WgS[(size_t)(n0 + ar) * HIDDEN + kk + ac * 8], &Bg[bo + ar * LDT + ac * 8]);
    cp16(&WuS[(size_t)(n0 + ar) * HIDDEN + kk + ac * 8], &Bu[bo + ar * LDT + ac * 8]);
  };

  stage(0, 0);
  int buf = 0;
  for (int kk = 0; kk < HIDDEN; kk += 32) {
    wait_stage();                 // my async chunks for `buf` have landed
    __syncthreads();              // everyone's landed; prior compute done
    if (kk + 32 < HIDDEN) stage(buf ^ 1, kk + 32);

    const int ao = buf * 128 * LDT, bo = buf * 64 * LDT;
    FragB a[2], bg[2], bu[2];
    #pragma unroll
    for (int i = 0; i < 2; ++i) {                  // A: lane=M, 16 contig K
      int m = wm * 32 + i * 16 + mrow;
      a[i].q[0] = *(const u32x4*)&As[ao + m * LDT + kb];
      a[i].q[1] = *(const u32x4*)&As[ao + m * LDT + kb + 8];
    }
    #pragma unroll
    for (int j = 0; j < 2; ++j) {                  // B: lane=N, 16 contig K
      int n = wn * 32 + j * 16 + mrow;
      bg[j].q[0] = *(const u32x4*)&Bg[bo + n * LDT + kb];
      bg[j].q[1] = *(const u32x4*)&Bg[bo + n * LDT + kb + 8];
      bu[j].q[0] = *(const u32x4*)&Bu[bo + n * LDT + kb];
      bu[j].q[1] = *(const u32x4*)&Bu[bo + n * LDT + kb + 8];
    }
    #pragma unroll
    for (int i = 0; i < 2; ++i)
      #pragma unroll
      for (int j = 0; j < 2; ++j) {
        ag[i][j].v = __builtin_amdgcn_wmma_f32_16x16x32_bf16(
            false, a[i].v, false, bg[j].v, (short)0, ag[i][j].v, false, false);
        au[i][j].v = __builtin_amdgcn_wmma_f32_16x16x32_bf16(
            false, a[i].v, false, bu[j].v, (short)0, au[i][j].v, false, false);
      }
    buf ^= 1;
  }
  // epilogue: t = silu(g) * u, store bf16.  C/D: lane=N, VGPR v -> M=v+8*(lane>=16)
  const int coln = lane & 15, rh = (lane >> 4) * 8;
  #pragma unroll
  for (int i = 0; i < 2; ++i)
    #pragma unroll
    for (int j = 0; j < 2; ++j)
      #pragma unroll
      for (int v = 0; v < 8; ++v) {
        float g = ag[i][j].f[v], u = au[i][j].f[v];
        float tv = (g / (1.f + __expf(-g))) * u;
        int row = m0 + wm * 32 + i * 16 + rh + v;
        int col = n0 + wn * 32 + j * 16 + coln;
        T[(size_t)row * KP + col] = f2bf(tv);
      }
}

// ----------------------------- down-proj GEMM ------------------------------
// Out[m][n] = sum_k T[m][k] * WdS[n][k].  M=SEQ, N=HIDDEN, K=KP.
__global__ __launch_bounds__(256) void k_down(
    const unsigned short* __restrict__ T,    // SEQ x KP bf16
    const unsigned short* __restrict__ WdS,  // HIDDEN x KP bf16 ([n][k])
    float* __restrict__ Out) {               // SEQ x HIDDEN fp32
  __shared__ __align__(16) unsigned short As[2 * 128 * LDT];
  __shared__ __align__(16) unsigned short Bs[2 * 64 * LDT];
  const int tid = threadIdx.x, lane = tid & 31, w = tid >> 5;
  const int wm = w & 3, wn = w >> 2;
  const int m0 = blockIdx.x * 128, n0 = blockIdx.y * 64;

  Acc acc[2][2];
  for (int i = 0; i < 2; ++i)
    for (int j = 0; j < 2; ++j)
      for (int e = 0; e < 8; ++e) acc[i][j].f[e] = 0.f;

  const int ar = tid >> 2, ac = tid & 3;
  const int mrow = lane & 15, kh = (lane >> 4) * 8, kb = (lane >> 4) * 16;

  auto stage = [&](int buf, int kk) {
    const int ao = buf * 128 * LDT, bo = buf * 64 * LDT;
    #pragma unroll
    for (int t2 = 0; t2 < 2; ++t2) {
      int r = ar + t2 * 64;
      cp16(&T[(size_t)(m0 + r) * KP + kk + ac * 8], &As[ao + r * LDT + ac * 8]);
    }
    cp16(&WdS[(size_t)(n0 + ar) * KP + kk + ac * 8], &Bs[bo + ar * LDT + ac * 8]);
  };

  stage(0, 0);
  int buf = 0;
  for (int kk = 0; kk < KP; kk += 32) {
    wait_stage();
    __syncthreads();
    if (kk + 32 < KP) stage(buf ^ 1, kk + 32);

    const int ao = buf * 128 * LDT, bo = buf * 64 * LDT;
    FragB a[2], b[2];
    #pragma unroll
    for (int i = 0; i < 2; ++i) {
      int m = wm * 32 + i * 16 + mrow;
      a[i].q[0] = *(const u32x4*)&As[ao + m * LDT + kb];
      a[i].q[1] = *(const u32x4*)&As[ao + m * LDT + kb + 8];
    }
    #pragma unroll
    for (int j = 0; j < 2; ++j) {
      int n = wn * 32 + j * 16 + mrow;
      b[j].q[0] = *(const u32x4*)&Bs[bo + n * LDT + kb];
      b[j].q[1] = *(const u32x4*)&Bs[bo + n * LDT + kb + 8];
    }
    #pragma unroll
    for (int i = 0; i < 2; ++i)
      #pragma unroll
      for (int j = 0; j < 2; ++j)
        acc[i][j].v = __builtin_amdgcn_wmma_f32_16x16x32_bf16(
            false, a[i].v, false, b[j].v, (short)0, acc[i][j].v, false, false);
    buf ^= 1;
  }
  const int coln = lane & 15, rh = (lane >> 4) * 8;
  #pragma unroll
  for (int i = 0; i < 2; ++i)
    #pragma unroll
    for (int j = 0; j < 2; ++j)
      #pragma unroll
      for (int v = 0; v < 8; ++v) {
        int row = m0 + wm * 32 + i * 16 + rh + v;
        int col = n0 + wn * 32 + j * 16 + coln;
        Out[(size_t)row * HIDDEN + col] = acc[i][j].f[v];
      }
}

// ------------------------------- launcher ----------------------------------
extern "C" void kernel_launch(void* const* d_in, const int* in_sizes, int n_in,
                              void* d_out, int out_size, void* d_ws, size_t ws_size,
                              hipStream_t stream) {
  const float* x  = (const float*)d_in[0];   // (1, 4096, 4096)
  const float* Wg = (const float*)d_in[1];   // (14336, 4096)
  const float* Wu = (const float*)d_in[2];   // (14336, 4096)
  const float* Wd = (const float*)d_in[3];   // (4096, 14336)
  const float* W1 = (const float*)d_in[4];   // (1024, 4096)
  const float* W2 = (const float*)d_in[5];   // (14336, 1024)
  float* out = (float*)d_out;                // (1, 4096, 4096) fp32

  // workspace carve (~59 MB), every piece 16B aligned
  char* wsp = (char*)d_ws;
  unsigned short* xb  = (unsigned short*)wsp; wsp += (size_t)SEQ * HIDDEN * 2;
  unsigned short* wgs = (unsigned short*)wsp; wsp += (size_t)KP * HIDDEN * 2;
  unsigned short* wus = (unsigned short*)wsp; wsp += (size_t)KP * HIDDEN * 2;
  unsigned short* wds = (unsigned short*)wsp; wsp += (size_t)HIDDEN * KP * 2;
  unsigned short* tb  = (unsigned short*)wsp; wsp += (size_t)SEQ * KP * 2;
  float* p  = (float*)wsp; wsp += (size_t)PRED_H * 4;
  float* h  = (float*)wsp; wsp += (size_t)INTER * 4;
  int*   idx = (int*)wsp;  wsp += (size_t)KP * 4;

  const float* x_last = x + (size_t)(SEQ - 1) * HIDDEN;

  k_pred1<<<PRED_H / 8, 256, 0, stream>>>(x_last, W1, p);        // 1024 waves
  k_pred2<<<INTER / 8, 256, 0, stream>>>(p, W2, h);              // 14336 waves
  k_topk<<<INTER / 256, 256, 0, stream>>>(h, idx);
  k_cvtx<<<2048, 256, 0, stream>>>(x, xb);
  k_gather_gu<<<KP, 256, 0, stream>>>(Wg, Wu, idx, wgs, wus);
  k_gather_d<<<HIDDEN, 256, 0, stream>>>(Wd, idx, wds);
  k_gateup<<<dim3(SEQ / 128, KP / 64), 256, 0, stream>>>(xb, wgs, wus, tb);
  k_down<<<dim3(SEQ / 128, HIDDEN / 64), 256, 0, stream>>>(tb, wds, out);
}
